// CrossBandWindowAttentionWav_30356828848466
// MI455X (gfx1250) — matrix operand order repaired and checked by reading, not compiled, not run on hardware
//
#include <hip/hip_runtime.h>
#include <hip/hip_bf16.h>

// ---- problem constants ----
#define NTOK 64      // tokens per window
#define CDIM 384     // channels
#define QKD  96      // q/k projected dim
#define NHEAD 6
#define HD   16      // q/k head dim
#define NWIN 1024    // mask windows
#define NB   2048    // batch*windows

typedef __attribute__((ext_vector_type(16))) __bf16 v16bf;
typedef __attribute__((ext_vector_type(8)))  __bf16 v8bf;
typedef __attribute__((ext_vector_type(4)))  __bf16 v4bf;
typedef __attribute__((ext_vector_type(8)))  float  v8f;

union FragU {
  v16bf v;
  v8bf  h[2];
  unsigned int w[8];
};

static __device__ __forceinline__ v8f wmma_bf16(v16bf a, v16bf b, v8f c) {
  return __builtin_amdgcn_wmma_f32_16x16x32_bf16(false, a, false, b, (short)0, c,
                                                 false, false);
}

// A-fragment (16x32 bf16) built from row-major f32 global memory, converting to bf16.
// Lane l<16: row m0+l, K = {half*8+0..7, 16+half*8+0..7} relative to kk.
static __device__ __forceinline__ v16bf load_a_f32g(const float* base, int ld,
                                                    int m0, int kk, int lane) {
  int row = lane & 15, half = lane >> 4;
  const float* p = base + (size_t)(m0 + row) * ld + kk + half * 8;
  float4 f0 = *(const float4*)(p);
  float4 f1 = *(const float4*)(p + 4);
  float4 f2 = *(const float4*)(p + 16);
  float4 f3 = *(const float4*)(p + 20);
  v16bf a;
  a[0]=(__bf16)f0.x; a[1]=(__bf16)f0.y; a[2]=(__bf16)f0.z; a[3]=(__bf16)f0.w;
  a[4]=(__bf16)f1.x; a[5]=(__bf16)f1.y; a[6]=(__bf16)f1.z; a[7]=(__bf16)f1.w;
  a[8]=(__bf16)f2.x; a[9]=(__bf16)f2.y; a[10]=(__bf16)f2.z; a[11]=(__bf16)f2.w;
  a[12]=(__bf16)f3.x; a[13]=(__bf16)f3.y; a[14]=(__bf16)f3.z; a[15]=(__bf16)f3.w;
  return a;
}

// B-fragment (32x16 bf16) from an N-major (transposed) bf16 weight: contiguous 32B.
// Lane l<16: col n0+l, K = {half*8+0..7, 16+half*8+0..7}.  Prefetch the next K-chunk (near-cache hint).
static __device__ __forceinline__ v16bf load_b_nmajor(const __bf16* Wt, int ldk,
                                                      int n0, int kk, int lane) {
  int ln = lane & 15, half = lane >> 4;
  const __bf16* p = Wt + (size_t)(n0 + ln) * ldk + kk + half * 8;
  __builtin_prefetch(p + 64, 0, 3);   // global_prefetch_b8, next chunk, near cache
  FragU u;
  u.h[0] = *(const v8bf*)(p);
  u.h[1] = *(const v8bf*)(p + 16);
  return u.v;
}

// A-fragment from row-major bf16 in LDS.
static __device__ __forceinline__ v16bf load_a_lds(const __bf16* base, int ld,
                                                   int m0, int kk, int lane) {
  int row = lane & 15, half = lane >> 4;
  const __bf16* p = base + (m0 + row) * ld + kk + half * 8;
  FragU u;
  u.h[0] = *(const v8bf*)(p);
  u.h[1] = *(const v8bf*)(p + 16);
  return u.v;
}

// ---------------- prep: weight transpose->bf16 + bias gather ----------------
// flat elems: Wqt 36864 | Wkt 36864 | Wvt 147456 | Wpt 147456 | bias 24576 = 393216
__global__ __launch_bounds__(256) void prep_kernel(
    const float* __restrict__ Wq, const float* __restrict__ Wk,
    const float* __restrict__ Wv, const float* __restrict__ Wp,
    const int* __restrict__ rpi, const float* __restrict__ rpb,
    __bf16* __restrict__ Wqt, __bf16* __restrict__ Wkt,
    __bf16* __restrict__ Wvt, __bf16* __restrict__ Wpt,
    float* __restrict__ biasW) {
  int i = blockIdx.x * 256 + threadIdx.x;
  if (i < 36864) {                       // Wqt[n][k] = Wq[k][n]
    int n = i / CDIM, k = i % CDIM;
    Wqt[i] = (__bf16)Wq[k * QKD + n];
  } else if (i < 73728) {
    int j = i - 36864; int n = j / CDIM, k = j % CDIM;
    Wkt[j] = (__bf16)Wk[k * QKD + n];
  } else if (i < 221184) {
    int j = i - 73728; int n = j / CDIM, k = j % CDIM;
    Wvt[j] = (__bf16)Wv[k * CDIM + n];
  } else if (i < 368640) {
    int j = i - 221184; int n = j / CDIM, k = j % CDIM;
    Wpt[j] = (__bf16)Wp[k * CDIM + n];
  } else if (i < 393216) {               // biasW[h][r][c] = rpb[rpi[r][c]][h]
    int j = i - 368640;
    int h = j / 4096, r = (j / 64) & 63, c = j & 63;
    biasW[j] = rpb[rpi[r * 64 + c] * NHEAD + h];
  }
}

// ---------------- fused per-window attention ----------------
// LDS (144 KB total => 2 workgroups per WGP):
//   qs[64][96] 12K | ks[64][96] 12K | vsT[384][64] 48K |
//   RB 48K  (cxs bf16 staging in stages 0-2, reused as os in stages 3c-4) |
//   atf[64][64] f32 16K | aps[64][64] bf16 8K
__global__ __launch_bounds__(256) void fused_attn_kernel(
    const float* __restrict__ x, const float* __restrict__ cx,
    const float* __restrict__ mask,
    const float* __restrict__ bq, const float* __restrict__ bk,
    const float* __restrict__ bv, const float* __restrict__ bp,
    const __bf16* __restrict__ Wqt, const __bf16* __restrict__ Wkt,
    const __bf16* __restrict__ Wvt, const __bf16* __restrict__ Wpt,
    const float* __restrict__ biasW, float* __restrict__ out) {
  extern __shared__ __align__(16) char smem[];
  __bf16* qs  = (__bf16*)smem;              // [64][96]
  __bf16* ks  = qs + NTOK * QKD;            // [64][96]
  __bf16* vsT = ks + NTOK * QKD;            // [384][64] feature-major V
  __bf16* RB  = vsT + CDIM * NTOK;          // 48K union: cxs / os
  float*  atf = (float*)(RB + NTOK * CDIM); // [64][64] f32 scores
  __bf16* aps = (__bf16*)(atf + NTOK * NTOK); // [64][64] probs
  __bf16* cxs = RB;                         // staged cross_x (bf16, row-major)
  __bf16* os  = RB;                         // attn output (bf16, row-major)

  const int b    = blockIdx.x;
  const int tid  = threadIdx.x;
  const int lane = tid & 31;
  const int wv   = tid >> 5;      // wave id 0..7
  const int ln   = lane & 15;
  const int half = lane >> 4;

  const float* xb   = x  + (size_t)b * NTOK * CDIM;
  const float* cxb  = cx + (size_t)b * NTOK * CDIM;
  const float* mrow = mask + (size_t)(b & (NWIN - 1)) * NTOK * NTOK;

  // ---- Stage 0: stage cross_x window into LDS as bf16 (read f32 once) ----
  for (int i = tid * 4; i < NTOK * CDIM; i += 256 * 4) {
    float4 f = *(const float4*)(cxb + i);
    v4bf t;
    t[0] = (__bf16)f.x; t[1] = (__bf16)f.y;
    t[2] = (__bf16)f.z; t[3] = (__bf16)f.w;
    *(v4bf*)(cxs + i) = t;
  }
  __syncthreads();

  // ---- Stage 1a: Q = (x@Wq + bq)*0.25 -> qs (A direct from global f32) ----
  for (int t = wv; t < 24; t += 8) {
    int m0 = (t / 6) * 16, n0 = (t % 6) * 16;
    v8f acc = {};
    for (int kk = 0; kk < CDIM; kk += 32) {
      v16bf a  = load_a_f32g(xb, CDIM, m0, kk, lane);
      v16bf bb = load_b_nmajor(Wqt, CDIM, n0, kk, lane);
      acc = wmma_bf16(a, bb, acc);
    }
    float bsc = bq[n0 + ln];
    for (int r = 0; r < 8; ++r)      // HEAD_DIM^-0.5 = 0.25 folded into q
      qs[(m0 + half * 8 + r) * QKD + n0 + ln] = (__bf16)((acc[r] + bsc) * 0.25f);
  }

  // ---- Stage 1b: K = cxs@Wk + bk -> ks (A from staged LDS bf16) ----
  for (int t = wv; t < 24; t += 8) {
    int m0 = (t / 6) * 16, n0 = (t % 6) * 16;
    v8f acc = {};
    for (int kk = 0; kk < CDIM; kk += 32) {
      v16bf a  = load_a_lds(cxs, CDIM, m0, kk, lane);
      v16bf bb = load_b_nmajor(Wkt, CDIM, n0, kk, lane);
      acc = wmma_bf16(a, bb, acc);
    }
    float bsc = bk[n0 + ln];
    for (int r = 0; r < 8; ++r)
      ks[(m0 + half * 8 + r) * QKD + n0 + ln] = (__bf16)(acc[r] + bsc);
  }

  // ---- Stage 2: V = cxs@Wv + bv -> LDS feature-major (vsT[feat][token]) ----
  for (int t = wv; t < 96; t += 8) {
    int m0 = (t / 24) * 16, n0 = (t % 24) * 16;
    v8f acc = {};
    for (int kk = 0; kk < CDIM; kk += 32) {
      v16bf a  = load_a_lds(cxs, CDIM, m0, kk, lane);
      v16bf bb = load_b_nmajor(Wvt, CDIM, n0, kk, lane);
      acc = wmma_bf16(a, bb, acc);
    }
    float bsc = bv[n0 + ln];
    for (int r = 0; r < 8; ++r)
      vsT[(n0 + ln) * NTOK + (m0 + half * 8 + r)] = (__bf16)(acc[r] + bsc);
  }
  __syncthreads();   // cxs dead from here; RB becomes os

  // ---- Stage 3: per head: S = q k^T + bias + mask ; softmax ; O_h = P V_h ----
  for (int h = 0; h < NHEAD; ++h) {
    // 3a: scores (K=16, zero-padded to one 16x16x32 WMMA per tile)
    for (int t = wv; t < 16; t += 8) {
      int m0 = (t / 4) * 16, n0 = (t % 4) * 16;
      FragU ua;
      ua.h[0] = *(const v8bf*)(qs + (m0 + ln) * QKD + h * HD + half * 8);
      ua.w[4] = 0; ua.w[5] = 0; ua.w[6] = 0; ua.w[7] = 0;   // K=16..31 pad
      FragU ub;
      ub.h[0] = *(const v8bf*)(ks + (n0 + ln) * QKD + h * HD + half * 8);
      ub.w[4] = 0; ub.w[5] = 0; ub.w[6] = 0; ub.w[7] = 0;   // K=16..31 pad
      v8f acc = {};
      acc = wmma_bf16(ua.v, ub.v, acc);
      for (int r = 0; r < 8; ++r) {
        int i = m0 + half * 8 + r, j = n0 + ln;
        atf[i * 64 + j] = acc[r] + biasW[h * 4096 + i * 64 + j] + mrow[i * 64 + j];
      }
    }
    __syncthreads();
    // 3b: row softmax (one row per thread, threads 0..63)
    if (tid < NTOK) {
      float mx = -3.4e38f;
      for (int j = 0; j < 64; ++j) mx = fmaxf(mx, atf[tid * 64 + j]);
      float s = 0.f;
      for (int j = 0; j < 64; ++j) {
        float e = __expf(atf[tid * 64 + j] - mx);
        atf[tid * 64 + j] = e; s += e;
      }
      float inv = 1.f / s;
      for (int j = 0; j < 64; ++j)
        aps[tid * 64 + j] = (__bf16)(atf[tid * 64 + j] * inv);
    }
    __syncthreads();
    // 3c: O_h = P(64x64) @ V_h(64x64), heads disjoint in feature dim
    for (int t = wv; t < 16; t += 8) {
      int m0 = (t / 4) * 16, n0 = (t % 4) * 16;
      v8f acc = {};
      for (int kk = 0; kk < 64; kk += 32) {
        v16bf a = load_a_lds(aps, 64, m0, kk, lane);
        FragU ub;
        const __bf16* p = vsT + (h * 64 + n0 + ln) * NTOK + kk + half * 8;
        ub.h[0] = *(const v8bf*)(p);
        ub.h[1] = *(const v8bf*)(p + 16);
        acc = wmma_bf16(a, ub.v, acc);
      }
      for (int r = 0; r < 8; ++r)
        os[(m0 + half * 8 + r) * CDIM + h * 64 + n0 + ln] = (__bf16)acc[r];
    }
    __syncthreads();
  }

  // ---- Stage 4: out = O @ Wp + bp -> global f32 ----
  float* ob = out + (size_t)b * NTOK * CDIM;
  for (int t = wv; t < 96; t += 8) {
    int m0 = (t / 24) * 16, n0 = (t % 24) * 16;
    v8f acc = {};
    for (int kk = 0; kk < CDIM; kk += 32) {
      v16bf a  = load_a_lds(os, CDIM, m0, kk, lane);
      v16bf bb = load_b_nmajor(Wpt, CDIM, n0, kk, lane);
      acc = wmma_bf16(a, bb, acc);
    }
    float bsc = bp[n0 + ln];
    for (int r = 0; r < 8; ++r)
      ob[(size_t)(m0 + half * 8 + r) * CDIM + n0 + ln] = acc[r] + bsc;
  }
}

extern "C" void kernel_launch(void* const* d_in, const int* in_sizes, int n_in,
                              void* d_out, int out_size, void* d_ws, size_t ws_size,
                              hipStream_t stream) {
  (void)in_sizes; (void)n_in; (void)out_size; (void)ws_size;
  const float* x    = (const float*)d_in[0];
  const float* cxp  = (const float*)d_in[1];
  const int*   rpi  = (const int*)d_in[2];
  const float* mask = (const float*)d_in[3];
  const float* Wq   = (const float*)d_in[4];
  const float* bq   = (const float*)d_in[5];
  const float* Wk   = (const float*)d_in[6];
  const float* bk   = (const float*)d_in[7];
  const float* Wv   = (const float*)d_in[8];
  const float* bv   = (const float*)d_in[9];
  const float* Wp   = (const float*)d_in[10];
  const float* bp   = (const float*)d_in[11];
  const float* rpb  = (const float*)d_in[12];

  // workspace layout: Wqt | Wkt | Wvt | Wpt (bf16, N-major) | biasW (f32)
  __bf16* Wqt = (__bf16*)d_ws;
  __bf16* Wkt = Wqt + QKD * CDIM;
  __bf16* Wvt = Wkt + QKD * CDIM;
  __bf16* Wpt = Wvt + CDIM * CDIM;
  float* biasW = (float*)(Wpt + CDIM * CDIM);

  prep_kernel<<<1536, 256, 0, stream>>>(Wq, Wk, Wv, Wp, rpi, rpb,
                                        Wqt, Wkt, Wvt, Wpt, biasW);

  size_t lds = (size_t)(NTOK * QKD * 2 + CDIM * NTOK + NTOK * CDIM + NTOK * NTOK) *
                   sizeof(__bf16) +
               (size_t)NTOK * NTOK * sizeof(float);  // 147456 B
  fused_attn_kernel<<<NB, 256, lds, stream>>>(x, cxp, mask, bq, bk, bv, bp,
                                              Wqt, Wkt, Wvt, Wpt, biasW,
                                              (float*)d_out);
}
